// SetUpConvLayer_55387898249840
// MI455X (gfx1250) — compile-verified, matrix operand order checked by reading, not run on hardware
//
#include <hip/hip_runtime.h>
#include <hip/hip_bf16.h>

// ---------------- problem constants (from reference) ----------------
#define N_SRC 8192
#define N_TGT 16384
#define DF    64          // feature dim
#define HF    128         // hidden dim
#define KNBR  32          // max neighbors
#define R2F   (0.12f * 0.12f)
#define KPAD  96          // 67 (DF+3) padded to 3*32 for K-dim of WMMA
#define MSGS  KPAD        // message-row stride in elements (192 B, 16B aligned)

typedef __attribute__((ext_vector_type(16))) __bf16       v16bf;
typedef __attribute__((ext_vector_type(8)))  float        v8f;
typedef __attribute__((ext_vector_type(4)))  unsigned int u32x4;

union FragAB { v16bf v; u32x4 q[2]; };
union FragC  { v8f  v; float f[8]; };

// -------------------------------------------------------------------
// Kernel 1: convert + swizzle weights into per-lane WMMA B-fragment
// layout. 16-bit operand per-lane layout (mirror of documented A
// 16x32 layout, with N on lanes):
//   lane L: n = 16*nt + (L & 15), khalf = (L>>4)*8
//   element e (=2v+hi): K = 32*kk + ((v&4)?16:0) + khalf + 2*(v&3) + hi
// W1sw: [kk=3][nt=8][lane=32][e=16] bf16 ; W2sw: [kk=4][nt=8][lane=32][e=16]
// -------------------------------------------------------------------
__global__ void prep_weights_kernel(const float* __restrict__ W1,
                                    const float* __restrict__ W2,
                                    __bf16* __restrict__ W1sw,
                                    __bf16* __restrict__ W2sw) {
  int tid  = blockIdx.x * blockDim.x + threadIdx.x;
  int nthr = gridDim.x * blockDim.x;
  for (int i = tid; i < 3 * 8 * 32 * 16; i += nthr) {
    int e = i & 15, lane = (i >> 4) & 31, nt = (i >> 9) & 7, kk = i >> 12;
    int v = e >> 1, hi = e & 1;
    int k = kk * 32 + ((v & 4) ? 16 : 0) + ((lane >> 4) << 3) + ((v & 3) << 1) + hi;
    int n = nt * 16 + (lane & 15);
    W1sw[i] = (__bf16)((k < DF + 3) ? W1[k * HF + n] : 0.0f);
  }
  for (int i = tid; i < 4 * 8 * 32 * 16; i += nthr) {
    int e = i & 15, lane = (i >> 4) & 31, nt = (i >> 9) & 7, kk = i >> 12;
    int v = e >> 1, hi = e & 1;
    int k = kk * 32 + ((v & 4) ? 16 : 0) + ((lane >> 4) << 3) + ((v & 3) << 1) + hi;
    int n = nt * 16 + (lane & 15);
    W2sw[i] = (__bf16)W2[k * HF + n];
  }
}

// -------------------------------------------------------------------
// Kernel 2: ball query. One thread per target; stream src positions
// through LDS in 256-point tiles; keep the 32 nearest-in-radius in a
// sorted register list (unrolled bubble insert -> v_cndmask chains,
// no scratch). Also emits the tgt_pos / tgt_batch tail of the output.
// (batch vectors are all-zero in this problem, so the batch-equality
// predicate of torch_cluster.radius is vacuously true.)
// -------------------------------------------------------------------
__global__ __launch_bounds__(256) void ball_query_kernel(
    const float* __restrict__ src_pos, const float* __restrict__ tgt_pos,
    const int* __restrict__ tgt_batch, int* __restrict__ nbr_idx,
    float* __restrict__ out_pos, int* __restrict__ out_batch) {
  __shared__ float sp[256 * 3];
  const int t = blockIdx.x * 256 + threadIdx.x;
  const float px = tgt_pos[t * 3 + 0];
  const float py = tgt_pos[t * 3 + 1];
  const float pz = tgt_pos[t * 3 + 2];

  float dist[KNBR];
  int   sid[KNBR];
#pragma unroll
  for (int k = 0; k < KNBR; ++k) { dist[k] = 3.0e38f; sid[k] = -1; }

  for (int base = 0; base < N_SRC; base += 256) {
    __syncthreads();
    const int s = base + threadIdx.x;
    sp[threadIdx.x * 3 + 0] = src_pos[s * 3 + 0];
    sp[threadIdx.x * 3 + 1] = src_pos[s * 3 + 1];
    sp[threadIdx.x * 3 + 2] = src_pos[s * 3 + 2];
    __syncthreads();
    for (int j = 0; j < 256; ++j) {
      const float dx = sp[j * 3 + 0] - px;
      const float dy = sp[j * 3 + 1] - py;
      const float dz = sp[j * 3 + 2] - pz;
      const float d2 = dx * dx + dy * dy + dz * dz;
      if (d2 <= R2F && d2 < dist[KNBR - 1]) {
        dist[KNBR - 1] = d2;
        sid[KNBR - 1]  = base + j;
#pragma unroll
        for (int k = KNBR - 1; k > 0; --k) {
          const float dk = dist[k], dk1 = dist[k - 1];
          const int   ik = sid[k],  ik1 = sid[k - 1];
          const bool  sw = dk < dk1;
          dist[k]     = sw ? dk1 : dk;
          dist[k - 1] = sw ? dk  : dk1;
          sid[k]      = sw ? ik1 : ik;
          sid[k - 1]  = sw ? ik  : ik1;
        }
      }
    }
  }
#pragma unroll
  for (int k = 0; k < KNBR; ++k) nbr_idx[t * KNBR + k] = sid[k];
  out_pos[t * 3 + 0] = px;
  out_pos[t * 3 + 1] = py;
  out_pos[t * 3 + 2] = pz;
  out_batch[t] = tgt_batch[t];
}

// -------------------------------------------------------------------
// Kernel 3: PointConv MLP + masked max, WMMA bf16 path.
// One workgroup (4 wave32) per target. Wave w owns output columns
// [32w, 32w+32). A-fragments (neighbor rows) are hoisted into
// registers once per layer and reused across both N-tiles, halving
// hot-loop LDS traffic vs. the naive schedule.
// -------------------------------------------------------------------
__global__ __launch_bounds__(128) void pointconv_kernel(
    const float* __restrict__ src_features, const float* __restrict__ src_pos,
    const float* __restrict__ tgt_pos, const int* __restrict__ nbr_idx,
    const __bf16* __restrict__ W1sw, const __bf16* __restrict__ W2sw,
    const float* __restrict__ b1, const float* __restrict__ b2,
    float* __restrict__ out) {
  __shared__ __attribute__((aligned(16))) __bf16 Msg[KNBR * MSGS];  // 32 x 96
  __shared__ __attribute__((aligned(16))) __bf16 H1s[KNBR * HF];    // 32 x 128
  __shared__ int rvalid[KNBR];

  const int t    = blockIdx.x;
  const int tid  = threadIdx.x;
  const int lane = tid & 31;
  const int wave = tid >> 5;
  const int l15  = lane & 15;
  const int kh   = (lane >> 4) << 3;  // 0 or 8 (K-half / M-half selector)

  __builtin_prefetch(W2sw + (lane << 4), 0, 3);  // global_prefetch_b8

  // ---- phase 0: build message matrix [32 x 96] bf16 in LDS ----
  if (tid < KNBR) rvalid[tid] = (nbr_idx[t * KNBR + tid] >= 0) ? 1 : 0;
  const float tx = tgt_pos[t * 3 + 0];
  const float ty = tgt_pos[t * 3 + 1];
  const float tz = tgt_pos[t * 3 + 2];
  for (int i = tid; i < KNBR * MSGS; i += 128) {
    const int r = i / MSGS;
    const int c = i - r * MSGS;
    const int s = nbr_idx[t * KNBR + r];
    float v = 0.0f;
    if (s >= 0) {
      if (c < DF)           v = src_features[s * DF + c];
      else if (c == DF)     v = src_pos[s * 3 + 0] - tx;
      else if (c == DF + 1) v = src_pos[s * 3 + 1] - ty;
      else if (c == DF + 2) v = src_pos[s * 3 + 2] - tz;
    }
    Msg[i] = (__bf16)v;
  }
  __syncthreads();

  // ---- layer 1: H1 = relu(Msg @ W1 + b1), K = 96 (3 wmma steps) ----
  {
    FragAB a1[2][3];  // A-fragments: invariant across N-tiles -> load once
#pragma unroll
    for (int mt = 0; mt < 2; ++mt) {
      const int m = mt * 16 + l15;
#pragma unroll
      for (int kk = 0; kk < 3; ++kk) {
        a1[mt][kk].q[0] = *(const u32x4*)(&Msg[m * MSGS + kk * 32 + kh]);
        a1[mt][kk].q[1] = *(const u32x4*)(&Msg[m * MSGS + kk * 32 + 16 + kh]);
      }
    }
#pragma unroll
    for (int nti = 0; nti < 2; ++nti) {
      const int nt  = wave * 2 + nti;
      const int col = nt * 16 + l15;
      const float bc = b1[col];
      FragAB bf[3];
#pragma unroll
      for (int kk = 0; kk < 3; ++kk) {
        const __bf16* wp = W1sw + (((kk * 8 + nt) * 32 + lane) << 4);
        bf[kk].q[0] = *(const u32x4*)(wp);
        bf[kk].q[1] = *(const u32x4*)(wp + 8);
      }
#pragma unroll
      for (int mt = 0; mt < 2; ++mt) {
        FragC acc;
#pragma unroll
        for (int r = 0; r < 8; ++r) acc.f[r] = 0.0f;
#pragma unroll
        for (int kk = 0; kk < 3; ++kk) {
          acc.v = __builtin_amdgcn_wmma_f32_16x16x32_bf16(
              false, a1[mt][kk].v, false, bf[kk].v, (short)0, acc.v, false, false);
        }
#pragma unroll
        for (int r = 0; r < 8; ++r) {
          float h = acc.f[r] + bc;
          h = h > 0.0f ? h : 0.0f;
          const int mrow = mt * 16 + kh + r;  // C/D layout: lanes>=16 hold M+8
          H1s[mrow * HF + col] = (__bf16)h;
        }
      }
    }
  }
  __syncthreads();

  // ---- layer 2: out = max_rows relu(H1 @ W2 + b2), K = 128 ----
  const int hasN = rvalid[0];  // sorted insert fills slot 0 first
  {
    FragAB a2[2][4];  // hoisted A-fragments (H1 rows), reused across N-tiles
#pragma unroll
    for (int mt = 0; mt < 2; ++mt) {
      const int m = mt * 16 + l15;
#pragma unroll
      for (int kk = 0; kk < 4; ++kk) {
        a2[mt][kk].q[0] = *(const u32x4*)(&H1s[m * HF + kk * 32 + kh]);
        a2[mt][kk].q[1] = *(const u32x4*)(&H1s[m * HF + kk * 32 + 16 + kh]);
      }
    }
#pragma unroll
    for (int nti = 0; nti < 2; ++nti) {
      const int nt  = wave * 2 + nti;
      const int col = nt * 16 + l15;
      const float bc = b2[col];
      FragAB bf[4];
#pragma unroll
      for (int kk = 0; kk < 4; ++kk) {
        const __bf16* wp = W2sw + (((kk * 8 + nt) * 32 + lane) << 4);
        bf[kk].q[0] = *(const u32x4*)(wp);
        bf[kk].q[1] = *(const u32x4*)(wp + 8);
      }
      float cmax = -3.0e38f;
#pragma unroll
      for (int mt = 0; mt < 2; ++mt) {
        FragC acc;
#pragma unroll
        for (int r = 0; r < 8; ++r) acc.f[r] = 0.0f;
#pragma unroll
        for (int kk = 0; kk < 4; ++kk) {
          acc.v = __builtin_amdgcn_wmma_f32_16x16x32_bf16(
              false, a2[mt][kk].v, false, bf[kk].v, (short)0, acc.v, false, false);
        }
#pragma unroll
        for (int r = 0; r < 8; ++r) {
          const int mrow = mt * 16 + kh + r;
          float h = acc.f[r] + bc;
          h = h > 0.0f ? h : 0.0f;
          h = rvalid[mrow] ? h : -3.0e38f;
          cmax = h > cmax ? h : cmax;
        }
      }
      // combine the two M-halves of the C tile (lane L <-> L+16, same N)
      const float other = __shfl_xor(cmax, 16, 32);
      cmax = other > cmax ? other : cmax;
      if (lane < 16) out[t * HF + col] = hasN ? cmax : 0.0f;
    }
  }
}

// -------------------------------------------------------------------
extern "C" void kernel_launch(void* const* d_in, const int* in_sizes, int n_in,
                              void* d_out, int out_size, void* d_ws, size_t ws_size,
                              hipStream_t stream) {
  const float* src_features = (const float*)d_in[0];
  const float* src_pos      = (const float*)d_in[1];
  // d_in[2] = src_batch (all zeros, unused)
  // d_in[3] = tgt_features (unused by reference)
  const float* tgt_pos      = (const float*)d_in[4];
  const int*   tgt_batch    = (const int*)d_in[5];
  const float* W1           = (const float*)d_in[6];
  const float* b1           = (const float*)d_in[7];
  const float* W2           = (const float*)d_in[8];
  const float* b2           = (const float*)d_in[9];

  // workspace layout (16B-aligned slabs)
  char* ws = (char*)d_ws;
  int*    nbr_idx = (int*)ws;                                   // 16384*32*4 = 2 MB
  __bf16* W1sw    = (__bf16*)(ws + (size_t)N_TGT * KNBR * 4);   // 24 KB
  __bf16* W2sw    = W1sw + 3 * 8 * 32 * 16;                     // 32 KB

  // output layout: out[16384*128] fp32, tgt_pos copy[16384*3], tgt_batch bits[16384]
  float* out       = (float*)d_out;
  float* out_pos   = out + (size_t)N_TGT * HF;
  int*   out_batch = (int*)(out_pos + (size_t)N_TGT * 3);

  prep_weights_kernel<<<24, 256, 0, stream>>>(W1, W2, W1sw, W2sw);
  ball_query_kernel<<<N_TGT / 256, 256, 0, stream>>>(src_pos, tgt_pos, tgt_batch,
                                                     nbr_idx, out_pos, out_batch);
  pointconv_kernel<<<N_TGT, 128, 0, stream>>>(src_features, src_pos, tgt_pos,
                                              nbr_idx, W1sw, W2sw, b1, b2, out);
}
